// Interaction_41412074668752
// MI455X (gfx1250) — matrix operand order repaired and checked by reading, not compiled, not run on hardware
//
#include <hip/hip_runtime.h>

typedef __attribute__((ext_vector_type(16))) __bf16 v16bf;
typedef __attribute__((ext_vector_type(8)))  float  v8f;

#define N_   512
#define C_   256
#define JT   32      // edges per j-tile (two 16-row WMMA M-tiles share each B fragment)
#define ESTR 264     // bf16 elements per e-row in LDS (16B-aligned row stride, bank-skewed)
#define FSTR 260     // floats per v/q/ve row in LDS (16B-aligned stride: 1040B, bank-skewed)

struct U8 { uint4 a, b; };

static __device__ __forceinline__ v16bf as_v16bf(uint4 a, uint4 b) {
  U8 u; u.a = a; u.b = b;
  return __builtin_bit_cast(v16bf, u);
}

// round-to-nearest-even f32 -> bf16 (as raw bits)
static __device__ __forceinline__ unsigned int f2bf1(float f) {
  unsigned int u = __float_as_uint(f);
  u = u + 0x7FFFu + ((u >> 16) & 1u);
  return u >> 16;
}
static __device__ __forceinline__ unsigned int f2bf2(float lo, float hi) {
  return f2bf1(lo) | (f2bf1(hi) << 16);
}

// low 32 bits of a generic pointer into the LDS aperture == LDS byte offset
static __device__ __forceinline__ unsigned int lds_addr(const void* p) {
  return (unsigned int)(unsigned long long)p;
}

// ---------------------------------------------------------------------------
// Pack Wq|Wk|Wv (each 256x256 f32, row-major, y = x @ W^T) into bf16 WMMA-B
// fragments. Slot layout: idx = ((nt*8 + kt)*32 + lane)*8 + vg, one uint
// (bf16 pair) per slot. B (32x16, KxN): lane<16 -> N=lane, K=kt*32+2*vg(+1);
// lane>=16 -> N=lane-16, K=kt*32+16+2*vg(+1). nt 0-15 -> Wq, 16-31 -> Wk,
// 32-47 -> Wv; d = (nt&15)*16 + N; B[k][n] = W[d][k].
// ---------------------------------------------------------------------------
__global__ __launch_bounds__(256) void pack_w_kernel(
    const float* __restrict__ Wq, const float* __restrict__ Wk,
    const float* __restrict__ Wv, unsigned int* __restrict__ out) {
  int idx  = blockIdx.x * 256 + threadIdx.x;          // 48*8*32*8 = 98304
  int vg   = idx & 7;
  int lane = (idx >> 3) & 31;
  int kt   = (idx >> 8) & 7;
  int nt   = idx >> 11;
  const float* W = (nt < 16) ? Wq : ((nt < 32) ? Wk : Wv);
  int d  = (nt & 15) * 16 + (lane & 15);
  int k0 = kt * 32 + ((lane & 16) ? 16 : 0) + 2 * vg;
  out[idx] = f2bf2(W[d * C_ + k0], W[d * C_ + k0 + 1]);
}

// ---------------------------------------------------------------------------
// Fused kernel: one block per row i. Flash-softmax over j in tiles of 32.
// ---------------------------------------------------------------------------
__global__ __launch_bounds__(256) void fused_edge_attn_kernel(
    const float* __restrict__ np_, const float* __restrict__ sp,
    const float* __restrict__ vp,  const unsigned int* __restrict__ wpk,
    const float* __restrict__ bq,  const float* __restrict__ bk,
    const float* __restrict__ bv,  float* __restrict__ out) {
  __shared__ __align__(16) unsigned short e_lds[JT * ESTR];  // bf16 A tiles
  __shared__ __align__(16) float v_lds[JT * FSTR];           // staged v tile
  __shared__ __align__(16) float q_lds[JT * FSTR];
  __shared__ __align__(16) float ve_lds[JT * FSTR];
  __shared__ __align__(16) float ni_lds[C_];
  __shared__ float bias_lds[3 * C_];
  __shared__ float acc_lds[C_];
  __shared__ float score_lds[JT];
  __shared__ float p_lds[JT];
  __shared__ float mls[3];  // running max, running sum, rescale

  const int t    = threadIdx.x;
  const int i    = blockIdx.x;
  const int lane = t & 31;
  const int w    = t >> 5;

  // ---- init ----
  ni_lds[t]            = np_[(size_t)i * C_ + t];
  bias_lds[t]          = bq[t];
  bias_lds[C_ + t]     = bk[t];
  bias_lds[2 * C_ + t] = bv[t];
  acc_lds[t] = 0.f;
  if (t == 0) { mls[0] = -__builtin_inff(); mls[1] = 0.f; }
  __syncthreads();

  for (int j0 = 0; j0 < N_; j0 += JT) {
    // ---- async stage of v tile (global -> LDS, no VGPR round trip) ----
    // 32 rows x 256 f32 = 8192 dwords; 256 threads x 8 iters x b128.
    {
      const int idx = t * 4;
#pragma unroll
      for (int it = 0; it < 8; ++it) {
        const int f   = it * 1024 + idx;   // dword index in tile
        const int row = f >> 8;
        const int col = f & 255;           // multiple of 4 -> 16B aligned
        unsigned int ldst = lds_addr(&v_lds[row * FSTR + col]);
        unsigned long long ga =
            (unsigned long long)(const void*)(vp + ((size_t)i * N_ + (j0 + row)) * C_ + col);
        asm volatile("global_load_async_to_lds_b128 %0, %1, off"
                     :: "v"(ldst), "v"(ga) : "memory");
      }
    }

    // ---- stage e = n_i * n_j * s  (bf16, WMMA-A row layout); zero scores ----
    {
      const int me = t >> 3;               // 0..31, 8 threads per row
      const size_t rowg = ((size_t)i * N_ + (j0 + me)) * C_;
      const size_t rown = (size_t)(j0 + me) * C_;
#pragma unroll
      for (int cc = 0; cc < 8; ++cc) {
        const int c0 = (t & 7) * 4 + cc * 32;  // 8 threads cover 128B contiguous
        float4 s4 = *(const float4*)(sp + rowg + c0);
        float4 nj = *(const float4*)(np_ + rown + c0);
        float4 ni = *(const float4*)(ni_lds + c0);
        unsigned int* dst = (unsigned int*)&e_lds[me * ESTR + c0];
        dst[0] = f2bf2(ni.x * nj.x * s4.x, ni.y * nj.y * s4.y);
        dst[1] = f2bf2(ni.z * nj.z * s4.z, ni.w * nj.w * s4.w);
      }
      if (t < JT) score_lds[t] = 0.f;
      // warm L2 for next tile's s rows (32 rows x 1KB, 128B stride per thread)
      if (j0 + JT < N_) {
        const int pr = t >> 3;
        const int pc = (t & 7) * 32;       // 32 floats = 128B stride
        __builtin_prefetch(sp + ((size_t)i * N_ + (j0 + JT + pr)) * C_ + pc, 0, 0);
      }
    }
    asm volatile("s_wait_asynccnt 0x0" ::: "memory");
    __syncthreads();

    // ---- preload A fragments for both M-tiles (8 K-tiles each) ----
    // A 16x32 bf16: lane<16 -> M=lane, K = kt*32 + {0..15};
    //               lane>=16 -> M=lane-16, K = kt*32 + {16..31}.
    v16bf afrag0[8], afrag1[8];
    {
      const int m    = lane & 15;
      const int koff = (lane & 16) ? 16 : 0;
#pragma unroll
      for (int kt = 0; kt < 8; ++kt) {
        const unsigned short* p0 = &e_lds[m * ESTR + kt * 32 + koff];
        const unsigned short* p1 = &e_lds[(16 + m) * ESTR + kt * 32 + koff];
        afrag0[kt] = as_v16bf(*(const uint4*)p0, *(const uint4*)(p0 + 8));
        afrag1[kt] = as_v16bf(*(const uint4*)p1, *(const uint4*)(p1 + 8));
      }
    }

    const int nloc  = lane & 15;
    const int mbase = (lane & 16) ? 8 : 0;

    // ---- q phase: nt = w, w+8 -> (y+bq)*v into q_lds ----
#pragma unroll
    for (int qq = 0; qq < 2; ++qq) {
      const int nt = w + qq * 8;
      v8f acc0 = {}, acc1 = {};
#pragma unroll
      for (int kt = 0; kt < 8; ++kt) {
        const uint4* bp = (const uint4*)(wpk + (((size_t)(nt * 8 + kt)) * 32 + lane) * 8);
        v16bf bfrag = as_v16bf(bp[0], bp[1]);
        acc0 = __builtin_amdgcn_wmma_f32_16x16x32_bf16(
            false, afrag0[kt], false, bfrag, (short)0, acc0, false, false);
        acc1 = __builtin_amdgcn_wmma_f32_16x16x32_bf16(
            false, afrag1[kt], false, bfrag, (short)0, acc1, false, false);
      }
      const int d = nt * 16 + nloc;
      const int c = d & (C_ - 1);
      const float bb = bias_lds[d];
      float vv0[8], vv1[8];
#pragma unroll
      for (int r = 0; r < 8; ++r) {
        vv0[r] = v_lds[(mbase + r) * FSTR + c];
        vv1[r] = v_lds[(16 + mbase + r) * FSTR + c];
      }
#pragma unroll
      for (int r = 0; r < 8; ++r) {
        q_lds[(mbase + r) * FSTR + c]      = (acc0[r] + bb) * vv0[r];
        q_lds[(16 + mbase + r) * FSTR + c] = (acc1[r] + bb) * vv1[r];
      }
    }
    __syncthreads();

    // ---- k phase: nt = w+16, w+24 -> score[m] += k_c * q_c (LDS atomics) ----
#pragma unroll
    for (int kk = 0; kk < 2; ++kk) {
      const int nt = w + 16 + kk * 8;
      v8f acc0 = {}, acc1 = {};
#pragma unroll
      for (int kt = 0; kt < 8; ++kt) {
        const uint4* bp = (const uint4*)(wpk + (((size_t)(nt * 8 + kt)) * 32 + lane) * 8);
        v16bf bfrag = as_v16bf(bp[0], bp[1]);
        acc0 = __builtin_amdgcn_wmma_f32_16x16x32_bf16(
            false, afrag0[kt], false, bfrag, (short)0, acc0, false, false);
        acc1 = __builtin_amdgcn_wmma_f32_16x16x32_bf16(
            false, afrag1[kt], false, bfrag, (short)0, acc1, false, false);
      }
      const int d = nt * 16 + nloc;
      const int c = d & (C_ - 1);
      const float bb = bias_lds[d];
      float vv0[8], vv1[8], qv0[8], qv1[8];
#pragma unroll
      for (int r = 0; r < 8; ++r) {
        vv0[r] = v_lds[(mbase + r) * FSTR + c];
        vv1[r] = v_lds[(16 + mbase + r) * FSTR + c];
        qv0[r] = q_lds[(mbase + r) * FSTR + c];
        qv1[r] = q_lds[(16 + mbase + r) * FSTR + c];
      }
#pragma unroll
      for (int r = 0; r < 8; ++r) {
        atomicAdd(&score_lds[mbase + r],      (acc0[r] + bb) * vv0[r] * qv0[r]);
        atomicAdd(&score_lds[16 + mbase + r], (acc1[r] + bb) * vv1[r] * qv1[r]);
      }
    }

    // ---- ve phase: nt = w+32, w+40 -> (y+bv)*v into ve_lds ----
#pragma unroll
    for (int vv2 = 0; vv2 < 2; ++vv2) {
      const int nt = w + 32 + vv2 * 8;
      v8f acc0 = {}, acc1 = {};
#pragma unroll
      for (int kt = 0; kt < 8; ++kt) {
        const uint4* bp = (const uint4*)(wpk + (((size_t)(nt * 8 + kt)) * 32 + lane) * 8);
        v16bf bfrag = as_v16bf(bp[0], bp[1]);
        acc0 = __builtin_amdgcn_wmma_f32_16x16x32_bf16(
            false, afrag0[kt], false, bfrag, (short)0, acc0, false, false);
        acc1 = __builtin_amdgcn_wmma_f32_16x16x32_bf16(
            false, afrag1[kt], false, bfrag, (short)0, acc1, false, false);
      }
      const int d = nt * 16 + nloc;
      const int c = d & (C_ - 1);
      const float bb = bias_lds[d];
      float vv0[8], vv1[8];
#pragma unroll
      for (int r = 0; r < 8; ++r) {
        vv0[r] = v_lds[(mbase + r) * FSTR + c];
        vv1[r] = v_lds[(16 + mbase + r) * FSTR + c];
      }
#pragma unroll
      for (int r = 0; r < 8; ++r) {
        ve_lds[(mbase + r) * FSTR + c]      = (acc0[r] + bb) * vv0[r];
        ve_lds[(16 + mbase + r) * FSTR + c] = (acc1[r] + bb) * vv1[r];
      }
    }
    __syncthreads();

    // ---- online softmax update (single thread; 32 edges) ----
    if (t == 0) {
      const float m0 = mls[0], l0 = mls[1];
      float mx = -__builtin_inff();
#pragma unroll
      for (int e = 0; e < JT; ++e) {
        const float sc = score_lds[e] * 0.0625f;  // / sqrt(256)
        score_lds[e] = sc;
        mx = fmaxf(mx, sc);
      }
      const float nm    = fmaxf(m0, mx);
      const float scale = __expf(m0 - nm);
      float sum = 0.f;
#pragma unroll
      for (int e = 0; e < JT; ++e) {
        const float pe = __expf(score_lds[e] - nm);
        p_lds[e] = pe;
        sum += pe;
      }
      mls[0] = nm; mls[1] = l0 * scale + sum; mls[2] = scale;
    }
    __syncthreads();

    // ---- acc[c] = acc[c]*scale + sum_e p[e]*ve[e][c] ----
    {
      float a = acc_lds[t] * mls[2];
#pragma unroll
      for (int e = 0; e < JT; ++e) a += p_lds[e] * ve_lds[e * FSTR + t];
      acc_lds[t] = a;
    }
    __syncthreads();
  }

  // ---- epilogue: x = acc/l + n_i ; normalize row ----
  const float x = acc_lds[t] / mls[1] + ni_lds[t];
  q_lds[t] = x * x;
  __syncthreads();
  if (t == 0) {
    float ssum = 0.f;
    for (int c = 0; c < C_; ++c) ssum += q_lds[c];
    mls[2] = rsqrtf(ssum);
  }
  __syncthreads();
  out[(size_t)i * C_ + t] = x * mls[2];
}

extern "C" void kernel_launch(void* const* d_in, const int* in_sizes, int n_in,
                              void* d_out, int out_size, void* d_ws, size_t ws_size,
                              hipStream_t stream) {
  const float* n  = (const float*)d_in[0];
  const float* s  = (const float*)d_in[1];
  const float* v  = (const float*)d_in[2];
  const float* Wq = (const float*)d_in[3];
  const float* bq = (const float*)d_in[4];
  const float* Wk = (const float*)d_in[5];
  const float* bk = (const float*)d_in[6];
  const float* Wv = (const float*)d_in[7];
  const float* bv = (const float*)d_in[8];
  unsigned int* wpk = (unsigned int*)d_ws;  // 48*8*32*8*4 = 384 KB packed bf16 weights

  pack_w_kernel<<<384, 256, 0, stream>>>(Wq, Wk, Wv, wpk);
  fused_edge_attn_kernel<<<N_, 256, 0, stream>>>(n, s, v, wpk, bq, bk, bv,
                                                 (float*)d_out);
}
